// SelfAttention_48266842472768
// MI455X (gfx1250) — matrix operand order repaired and checked by reading, not compiled, not run on hardware
//
#include <hip/hip_runtime.h>
#include <hip/hip_bf16.h>

// ---------------------------------------------------------------------------
// Types for CDNA5 WMMA (wave32): D(16x16 f32) = A(16x32 bf16) x B(32x16 bf16) + C
// ---------------------------------------------------------------------------
typedef __bf16 v16bf __attribute__((ext_vector_type(16)));
typedef float  v8f   __attribute__((ext_vector_type(8)));

union FragBf { v16bf v; unsigned int u[8]; };

__device__ __forceinline__ v8f wmma_bf16(const FragBf& a, const FragBf& b, v8f c) {
    return __builtin_amdgcn_wmma_f32_16x16x32_bf16(
        /*neg_a=*/false, a.v, /*neg_b=*/false, b.v,
        /*c_mod=*/(short)0, c, /*reuse_a=*/false, /*reuse_b=*/false);
}

// Generic pointer to a __shared__ object: low 32 bits are the LDS byte offset
// (aperture check in CDNA5 ISA).
__device__ __forceinline__ unsigned lds_addr32(const void* p) {
    return (unsigned)(uintptr_t)p;
}

// 32B async copy global->LDS per lane: two b128 transfers (INST_OFFSET applies
// to both global and LDS addresses per CDNA5 ISA).
__device__ __forceinline__ void async_copy_32B(unsigned lds, const void* gaddr) {
    asm volatile("global_load_async_to_lds_b128 %0, %1, off\n\t"
                 "global_load_async_to_lds_b128 %0, %1, off offset:16"
                 :: "v"(lds), "v"(gaddr) : "memory");
}
__device__ __forceinline__ void async_copy_16B(unsigned lds, const void* gaddr) {
    asm volatile("global_load_async_to_lds_b128 %0, %1, off"
                 :: "v"(lds), "v"(gaddr) : "memory");
}
__device__ __forceinline__ void wait_async0() {
    asm volatile("s_wait_asynccnt 0" ::: "memory");
}

// ---------------------------------------------------------------------------
// Problem constants
// ---------------------------------------------------------------------------
static constexpr int B_  = 4;
static constexpr int T_  = 2048;
static constexpr int C_  = 1024;
static constexpr int NH_ = 16;
static constexpr int HS_ = 64;          // head size
static constexpr int M_  = B_ * T_;     // 8192 rows
static constexpr int QKVC = 3 * C_;     // 3072

// ---------------------------------------------------------------------------
// fp32 -> bf16 conversion (grid-stride)
// ---------------------------------------------------------------------------
__global__ void cvt_f32_bf16(const float* __restrict__ in, __bf16* __restrict__ out, int n) {
    for (int i = blockIdx.x * blockDim.x + threadIdx.x; i < n; i += gridDim.x * blockDim.x)
        out[i] = (__bf16)in[i];
}

// ---------------------------------------------------------------------------
// GEMM + bias:  OUT[m,n] = sum_k A[m,k] * W[n,k] + bias[n]
// A: MxK bf16 row-major, W: NxK bf16 row-major (torch Linear layout)
// Block tile 128x128, 256 threads = 8 waves (2 M x 4 N), wave tile 64x32.
// Double-buffered LDS tiles staged with GLOBAL_LOAD_ASYNC_TO_LDS_B128:
// the async DMA for tile k+1 overlaps the WMMAs of tile k (ASYNCcnt drained
// only after the compute phase).
// ---------------------------------------------------------------------------
template <bool OUT_BF16>
__global__ __launch_bounds__(256) void gemm_bias_kernel(
    const __bf16* __restrict__ A, const __bf16* __restrict__ W,
    const float* __restrict__ bias, void* __restrict__ out,
    int M, int N, int K)
{
    __shared__ __bf16 As[2][128][32];   // [buf][m][k]
    __shared__ __bf16 Bs[2][128][32];   // [buf][n][k]

    const int tid   = threadIdx.x;
    const int wid   = tid >> 5;
    const int lane  = tid & 31;
    const int half  = lane >> 4;     // lane half (0/1)
    const int l15   = lane & 15;
    const int waveM = wid >> 2;      // 0..1
    const int waveN = wid & 3;       // 0..3
    const int mBase = blockIdx.y * 128;
    const int nBase = blockIdx.x * 128;

    v8f acc[4][2];
    #pragma unroll
    for (int mt = 0; mt < 4; ++mt)
        #pragma unroll
        for (int nt = 0; nt < 2; ++nt)
            acc[mt][nt] = (v8f)0.0f;

    const int ldRow = tid >> 1;              // 0..127
    const int ldCol = (tid & 1) * 16;        // 0 or 16
    const unsigned ldsA[2] = { lds_addr32(&As[0][ldRow][ldCol]), lds_addr32(&As[1][ldRow][ldCol]) };
    const unsigned ldsB[2] = { lds_addr32(&Bs[0][ldRow][ldCol]), lds_addr32(&Bs[1][ldRow][ldCol]) };
    const __bf16* gA = A + (size_t)(mBase + ldRow) * K + ldCol;
    const __bf16* gW = W + (size_t)(nBase + ldRow) * K + ldCol;

    // prologue: stage tile 0 into buffer 0
    async_copy_32B(ldsA[0], gA);
    async_copy_32B(ldsB[0], gW);
    wait_async0();
    __syncthreads();

    int buf = 0;
    for (int k0 = 0; k0 < K; k0 += 32, buf ^= 1) {
        // issue async DMA for the next tile into the other buffer; it runs
        // concurrently with the WMMA phase below.
        if (k0 + 32 < K) {
            async_copy_32B(ldsA[buf ^ 1], gA + k0 + 32);
            async_copy_32B(ldsB[buf ^ 1], gW + k0 + 32);
        }

        // A fragments (ISA 16-bit A 16x32 layout: per-half split K)
        FragBf af[4];
        #pragma unroll
        for (int mt = 0; mt < 4; ++mt) {
            const int ar = waveM * 64 + mt * 16 + l15;
            #pragma unroll
            for (int d = 0; d < 4; ++d) {
                af[mt].u[d]     = *(const unsigned int*)&As[buf][ar][half * 8 + 2 * d];
                af[mt].u[4 + d] = *(const unsigned int*)&As[buf][ar][16 + half * 8 + 2 * d];
            }
        }
        // B fragments (32x16: same per-half split K as A)
        FragBf bf_[2];
        #pragma unroll
        for (int nt = 0; nt < 2; ++nt) {
            const int br = waveN * 32 + nt * 16 + l15;
            #pragma unroll
            for (int i = 0; i < 8; ++i)
                bf_[nt].u[i] = *(const unsigned int*)&Bs[buf][br][half * 8 + (i & 3) * 2 + (i >> 2) * 16];
        }
        #pragma unroll
        for (int mt = 0; mt < 4; ++mt)
            #pragma unroll
            for (int nt = 0; nt < 2; ++nt)
                acc[mt][nt] = wmma_bf16(af[mt], bf_[nt], acc[mt][nt]);

        // drain this wave's async copies, then rendezvous: after the barrier
        // the next buffer is fully resident for every wave.
        wait_async0();
        __syncthreads();
    }

    // epilogue: C/D layout: lane l15 = col; VGPR i = row (i + 8*half)
    #pragma unroll
    for (int mt = 0; mt < 4; ++mt) {
        #pragma unroll
        for (int nt = 0; nt < 2; ++nt) {
            const int col = nBase + waveN * 32 + nt * 16 + l15;
            const float bv = bias[col];
            #pragma unroll
            for (int i = 0; i < 8; ++i) {
                const int row = mBase + waveM * 64 + mt * 16 + i + 8 * half;
                const float v = acc[mt][nt][i] + bv;
                if (OUT_BF16) ((__bf16*)out)[(size_t)row * N + col] = (__bf16)v;
                else          ((float*)out)[(size_t)row * N + col] = v;
            }
        }
    }
}

// ---------------------------------------------------------------------------
// Flash-attention (causal) over qkv buffer.
// grid: (T/64, NH, B); block 128 threads = 4 waves; each wave owns 16 q rows.
// Q tile 64x64, KV tiles 128 keys; online softmax; O accumulated in v8f regs.
// Q/K tiles staged with async global->LDS copies.
// ---------------------------------------------------------------------------
__global__ __launch_bounds__(128) void attn_kernel(
    const __bf16* __restrict__ qkv, __bf16* __restrict__ y)
{
    __shared__ __bf16 Qs[64][64];    // [q][d]
    __shared__ __bf16 Ks[128][64];   // [j][d]
    __shared__ __bf16 Vt[64][128];   // [d][j]  (transposed for B-frag reads)
    __shared__ __bf16 Ps[64][128];   // [q][j]

    const int tid  = threadIdx.x;
    const int wid  = tid >> 5;
    const int lane = tid & 31;
    const int half = lane >> 4;
    const int l15  = lane & 15;
    const int qIdx = blockIdx.x;
    const int h    = blockIdx.y;
    const int b    = blockIdx.z;

    const size_t rowBase = (size_t)b * T_;
    const int    qBase   = qIdx * 64;
    const size_t ld      = QKVC;
    const float  scale   = 0.125f; // hs^-0.5 = 1/8

    // load Q tile (64 rows x 64 cols) via async copies
    #pragma unroll
    for (int i = 0; i < 4; ++i) {
        const int idx = tid + i * 128;           // 0..511
        const int r = idx >> 3, seg = idx & 7;
        async_copy_16B(lds_addr32(&Qs[r][seg * 8]),
                       qkv + (rowBase + qBase + r) * ld + (size_t)h * HS_ + seg * 8);
    }

    float m_i[8], l_i[8];
    v8f o[4];
    #pragma unroll
    for (int i = 0; i < 8; ++i) { m_i[i] = -1e30f; l_i[i] = 0.0f; }
    #pragma unroll
    for (int nt = 0; nt < 4; ++nt) o[nt] = (v8f)0.0f;

    const int nkv = (qBase + 63) / 128 + 1;      // causal: skip fully-masked tiles
    for (int kv = 0; kv < nkv; ++kv) {
        const int kvBase = kv * 128;
        __syncthreads();   // previous iteration's Ks/Vt/Ps reads done

        // K tile: 128x64 row-major, async global->LDS
        #pragma unroll
        for (int i = 0; i < 8; ++i) {
            const int idx = tid + i * 128;       // 0..1023
            const int r = idx >> 3, seg = idx & 7;
            async_copy_16B(lds_addr32(&Ks[r][seg * 8]),
                           qkv + (rowBase + kvBase + r) * ld + C_ + (size_t)h * HS_ + seg * 8);
        }
        // V tile transposed: Vt[d][j] (manual transpose via registers),
        // overlaps with the async K copies above.
        #pragma unroll
        for (int i = 0; i < 8; ++i) {
            const int idx = tid + i * 128;
            const int r = idx >> 3, seg = idx & 7;
            uint4 vv = *(const uint4*)(qkv + (rowBase + kvBase + r) * ld + 2 * C_ + (size_t)h * HS_ + seg * 8);
            const __bf16* pv = (const __bf16*)&vv;
            #pragma unroll
            for (int j = 0; j < 8; ++j) Vt[seg * 8 + j][r] = pv[j];
        }
        wait_async0();     // Q (first iter) + K complete
        __syncthreads();

        // S = Q * K^T   (K-dim = hs = 64 -> 2 WMMA K-steps)
        v8f s[8];
        #pragma unroll
        for (int nt = 0; nt < 8; ++nt) s[nt] = (v8f)0.0f;
        #pragma unroll
        for (int ks = 0; ks < 2; ++ks) {
            FragBf a;
            const int ar = wid * 16 + l15;
            #pragma unroll
            for (int d = 0; d < 4; ++d) {
                a.u[d]     = *(const unsigned int*)&Qs[ar][ks * 32 + half * 8 + 2 * d];
                a.u[4 + d] = *(const unsigned int*)&Qs[ar][ks * 32 + 16 + half * 8 + 2 * d];
            }
            #pragma unroll
            for (int nt = 0; nt < 8; ++nt) {
                FragBf bfr;
                const int br = nt * 16 + l15;    // key index j
                #pragma unroll
                for (int i = 0; i < 8; ++i)
                    bfr.u[i] = *(const unsigned int*)&Ks[br][ks * 32 + half * 8 + (i & 3) * 2 + (i >> 2) * 16];
                s[nt] = wmma_bf16(a, bfr, s[nt]);
            }
        }

        // scale + causal mask
        #pragma unroll
        for (int nt = 0; nt < 8; ++nt) {
            const int j = kvBase + nt * 16 + l15;
            #pragma unroll
            for (int i = 0; i < 8; ++i) {
                const int q = qBase + wid * 16 + i + 8 * half;
                const float v = s[nt][i] * scale;
                s[nt][i] = (j <= q) ? v : -1e30f;
            }
        }

        // online softmax: row max (xor masks 1..8 stay inside 16-lane half)
        #pragma unroll
        for (int i = 0; i < 8; ++i) {
            float mx = s[0][i];
            #pragma unroll
            for (int nt = 1; nt < 8; ++nt) mx = fmaxf(mx, s[nt][i]);
            #pragma unroll
            for (int off = 1; off < 16; off <<= 1) mx = fmaxf(mx, __shfl_xor(mx, off, 32));
            const float mnew = fmaxf(m_i[i], mx);
            const float alpha = __expf(m_i[i] - mnew);
            l_i[i] *= alpha;
            #pragma unroll
            for (int nt = 0; nt < 4; ++nt) o[nt][i] *= alpha;
            m_i[i] = mnew;
        }
        // p = exp(s - m); row sums
        #pragma unroll
        for (int nt = 0; nt < 8; ++nt)
            #pragma unroll
            for (int i = 0; i < 8; ++i)
                s[nt][i] = __expf(s[nt][i] - m_i[i]);
        #pragma unroll
        for (int i = 0; i < 8; ++i) {
            float sm = 0.0f;
            #pragma unroll
            for (int nt = 0; nt < 8; ++nt) sm += s[nt][i];
            #pragma unroll
            for (int off = 1; off < 16; off <<= 1) sm += __shfl_xor(sm, off, 32);
            l_i[i] += sm;
        }
        // write P to LDS (re-layout C-form -> row-major for A-frag reads)
        #pragma unroll
        for (int nt = 0; nt < 8; ++nt) {
            const int colp = nt * 16 + l15;
            #pragma unroll
            for (int i = 0; i < 8; ++i)
                Ps[wid * 16 + i + 8 * half][colp] = (__bf16)s[nt][i];
        }
        __syncthreads();

        // O += P * V   (K-dim = 128 keys -> 4 WMMA K-steps)
        #pragma unroll
        for (int ks = 0; ks < 4; ++ks) {
            FragBf a;
            const int ar = wid * 16 + l15;
            #pragma unroll
            for (int d = 0; d < 4; ++d) {
                a.u[d]     = *(const unsigned int*)&Ps[ar][ks * 32 + half * 8 + 2 * d];
                a.u[4 + d] = *(const unsigned int*)&Ps[ar][ks * 32 + 16 + half * 8 + 2 * d];
            }
            #pragma unroll
            for (int nt = 0; nt < 4; ++nt) {
                FragBf bfr;
                const int br = nt * 16 + l15;    // output dim d
                #pragma unroll
                for (int i = 0; i < 8; ++i)
                    bfr.u[i] = *(const unsigned int*)&Vt[br][ks * 32 + half * 8 + (i & 3) * 2 + (i >> 2) * 16];
                o[nt] = wmma_bf16(a, bfr, o[nt]);
            }
        }
    }

    // normalize and store y (bf16) at [b*T + t, h*64 + d]
    #pragma unroll
    for (int nt = 0; nt < 4; ++nt) {
        const int d = nt * 16 + l15;
        #pragma unroll
        for (int i = 0; i < 8; ++i) {
            const int t = qBase + wid * 16 + i + 8 * half;
            const float val = o[nt][i] / l_i[i];
            y[(rowBase + t) * C_ + (size_t)h * HS_ + d] = (__bf16)val;
        }
    }
}

// ---------------------------------------------------------------------------
// Launch
// ---------------------------------------------------------------------------
extern "C" void kernel_launch(void* const* d_in, const int* in_sizes, int n_in,
                              void* d_out, int out_size, void* d_ws, size_t ws_size,
                              hipStream_t stream) {
    (void)in_sizes; (void)n_in; (void)out_size; (void)ws_size;
    const float* x      = (const float*)d_in[0];   // (B,T,C)
    const float* w_attn = (const float*)d_in[1];   // (3C,C)
    const float* b_attn = (const float*)d_in[2];   // (3C,)
    const float* w_proj = (const float*)d_in[3];   // (C,C)
    const float* b_proj = (const float*)d_in[4];   // (C,)
    float* out = (float*)d_out;                    // (B,T,C) fp32

    char* ws = (char*)d_ws;
    __bf16* xb   = (__bf16*)ws;                                ws += (size_t)M_ * C_ * 2;
    __bf16* wab  = (__bf16*)ws;                                ws += (size_t)QKVC * C_ * 2;
    __bf16* wpb  = (__bf16*)ws;                                ws += (size_t)C_ * C_ * 2;
    __bf16* qkvb = (__bf16*)ws;                                ws += (size_t)M_ * QKVC * 2;
    __bf16* yb   = (__bf16*)ws;

    // convert inputs to bf16
    cvt_f32_bf16<<<2048, 256, 0, stream>>>(x,      xb,  M_ * C_);
    cvt_f32_bf16<<<1024, 256, 0, stream>>>(w_attn, wab, QKVC * C_);
    cvt_f32_bf16<<<512,  256, 0, stream>>>(w_proj, wpb, C_ * C_);

    // qkv = x @ w_attn^T + b_attn   (8192 x 3072, K=1024) -> bf16
    gemm_bias_kernel<true><<<dim3(QKVC / 128, M_ / 128), 256, 0, stream>>>(
        xb, wab, b_attn, qkvb, M_, QKVC, C_);

    // causal flash attention -> y (bf16)
    attn_kernel<<<dim3(T_ / 64, NH_, B_), 128, 0, stream>>>(qkvb, yb);

    // out = y @ w_proj^T + b_proj   (8192 x 1024, K=1024) -> fp32
    gemm_bias_kernel<false><<<dim3(C_ / 128, M_ / 128), 256, 0, stream>>>(
        yb, wpb, b_proj, out, M_, C_, C_);
}
